// OrthogonalRandomAttention_5257039970889
// MI455X (gfx1250) — matrix operand order repaired and checked by reading, not compiled, not run on hardware
//
#include <hip/hip_runtime.h>

// ---------------------------------------------------------------------------
// Orthogonal random-feature linear attention for MI455X (gfx1250).
// - All matmuls via v_wmma_f32_16x16x32_bf16 (fp32 accumulate).
// - All GEMM B-operands stored pre-transposed (BT[N][K]) so A and B tiles are
//   identical [128][32] bf16 blocks -> staged with TENSOR_LOAD_TO_LDS (TDM),
//   double-buffered, fragments read as aligned ds_load_b128 pairs.
// ---------------------------------------------------------------------------

typedef __bf16 bf16_t;
typedef __attribute__((ext_vector_type(16))) __bf16 v16bf;
typedef __attribute__((ext_vector_type(8)))  float  v8f;
typedef __attribute__((ext_vector_type(4)))  unsigned uint32x4;
typedef __attribute__((ext_vector_type(8)))  int      int32x8;
typedef __attribute__((ext_vector_type(4)))  int      int32x4;

#define NB  4      // batch
#define NH  8      // heads
#define NS  4096   // seq
#define NDH 128    // head dim
#define ND  1024   // model dim
#define NM  256    // feature dim (sin||cos)

enum GemmMode { MODE_QKV = 0, MODE_QKVT = 1, MODE_PHI_Q = 2, MODE_PHI_K = 3,
                MODE_KVT = 4, MODE_OUT = 5, MODE_FINAL = 6 };

#if __has_builtin(__builtin_amdgcn_tensor_load_to_lds)
#define USE_TDM 1
#else
#define USE_TDM 0
#endif

__device__ __forceinline__ bf16_t f2bf(float f) {
  unsigned u = __float_as_uint(f);
  unsigned r = u + 0x7FFFu + ((u >> 16) & 1u);   // round-to-nearest-even
  unsigned short s = (unsigned short)(r >> 16);
  return __builtin_bit_cast(bf16_t, s);
}

// ------------------------------ fp32 -> bf16 -------------------------------
__global__ void cvt_kernel(const float* __restrict__ src,
                           bf16_t* __restrict__ dst, int n) {
  int i = blockIdx.x * blockDim.x + threadIdx.x;
  int stride = gridDim.x * blockDim.x;
  for (; i < n; i += stride) dst[i] = f2bf(src[i]);
}

// -------------------- fp32 [R][C] -> bf16 transposed [C][R] ----------------
__global__ __launch_bounds__(256)
void cvt_t_kernel(const float* __restrict__ src, bf16_t* __restrict__ dst,
                  int R, int C) {
  __shared__ float t[32][33];
  int c0 = blockIdx.x * 32, r0 = blockIdx.y * 32;
  int tx = threadIdx.x & 31, ty = threadIdx.x >> 5;   // ty in 0..7
#pragma unroll
  for (int i = ty; i < 32; i += 8) t[i][tx] = src[(long)(r0 + i) * C + c0 + tx];
  __syncthreads();
#pragma unroll
  for (int i = ty; i < 32; i += 8)
    dst[(long)(c0 + i) * R + r0 + tx] = f2bf(t[tx][i]);
}

// ----------------------- TDM tile load: [128 rows][32 k] -------------------
#if USE_TDM
__device__ __forceinline__ void wait_tensor0() {
#if __has_builtin(__builtin_amdgcn_s_wait_tensorcnt)
  __builtin_amdgcn_s_wait_tensorcnt(0);
#else
  asm volatile("s_wait_tensorcnt 0x0" ::: "memory");
#endif
}

// 2D tile load: 32 contiguous bf16 per row, 128 rows, row stride = ld elems.
__device__ __forceinline__ void tdm_load_tile(unsigned lds_addr,
                                              const bf16_t* gsrc,
                                              unsigned ld, unsigned kdim,
                                              unsigned rows) {
  unsigned long long ga = (unsigned long long)(const void*)gsrc;
  uint32x4 g0;
  g0[0] = 1u;                                            // count=1 (valid D#)
  g0[1] = lds_addr;                                      // LDS byte address
  g0[2] = (unsigned)ga;                                  // global_addr lo
  g0[3] = (unsigned)((ga >> 32) & 0x01FFFFFFull) | (2u << 30);  // hi | type=2
  int32x8 g1;
  g1[0] = (int)(1u << 16);                               // data_size = 2B
  g1[1] = (int)((kdim & 0xFFFFu) << 16);                 // tensor_dim0 lo16
  g1[2] = (int)((kdim >> 16) | ((rows & 0xFFFFu) << 16));// dim0 hi | dim1 lo
  g1[3] = (int)((rows >> 16) | (32u << 16));             // dim1 hi | tile0=32
  g1[4] = 128;                                           // tile_dim1=128
  g1[5] = (int)ld;                                       // dim0_stride lo32
  g1[6] = 0;
  g1[7] = 0;
  int32x4 z4 = {0, 0, 0, 0};
#if __clang_major__ >= 23
  int32x8 z8 = {0, 0, 0, 0, 0, 0, 0, 0};
  __builtin_amdgcn_tensor_load_to_lds(g0, g1, z4, z4, z8, 0);
#else
  __builtin_amdgcn_tensor_load_to_lds(g0, g1, z4, z4, 0);
#endif
}
#endif  // USE_TDM

// fallback per-thread tile copy (also used if TDM builtin is unavailable)
__device__ __forceinline__ void copy_tile(bf16_t (*dst)[32],
                                          const bf16_t* __restrict__ src,
                                          int ld, int tid) {
#pragma unroll
  for (int c = 0; c < 2; ++c) {
    int ch = tid + c * 256;
    int row = ch >> 2, cs = (ch & 3) * 8;
    *(uint4*)&dst[row][cs] = *(const uint4*)(src + (long)row * ld + cs);
  }
}

// ------------------------- generic bf16 WMMA GEMM --------------------------
// C[M,N] = A[M,K] (row-major, lda) x BT[N,K] (row-major, ldbt), batched over
// blockIdx.y. BM=BN=128, BK=32, double-buffered LDS staged via TDM.
// 256 thr = 8 waves; each wave owns a 32x64 sub-tile (2x4 WMMA accumulators).
__global__ __launch_bounds__(256)
void wmma_gemm_kernel(const bf16_t* __restrict__ A, const bf16_t* __restrict__ BT,
                      void* __restrict__ Out, const float* __restrict__ aux,
                      int K, int N, long sA, long sB, int lda, int ldbt, int mode)
{
  __shared__ bf16_t As[2][128][32];
  __shared__ bf16_t Bs[2][128][32];

  const int tid  = threadIdx.x;
  const int lane = tid & 31;
  const int wid  = tid >> 5;
  const int kh   = lane >> 4;       // which K-half this lane holds
  const int l15  = lane & 15;
  const int wm0  = (wid & 3) * 32;  // wave M offset inside block tile
  const int wn0  = (wid >> 2) * 64; // wave N offset inside block tile

  const int numN = N >> 7;
  const int bm0  = (blockIdx.x / numN) << 7;
  const int bn0  = (blockIdx.x % numN) << 7;

  const bf16_t* Ay = A  + (long)blockIdx.y * sA + (long)bm0 * lda;
  const bf16_t* By = BT + (long)blockIdx.y * sB + (long)bn0 * ldbt;

  v8f zero = {};
  v8f acc[2][4];
#pragma unroll
  for (int i = 0; i < 2; ++i)
#pragma unroll
    for (int j = 0; j < 4; ++j) acc[i][j] = zero;

  const int nk = K >> 5;

  // ---- prologue: stage tile 0 ----
#if USE_TDM
  if (wid == 0)
    tdm_load_tile((unsigned)(unsigned long long)(const void*)&As[0][0][0],
                  Ay, (unsigned)lda, (unsigned)K, 1u << 20);
  else if (wid == 1)
    tdm_load_tile((unsigned)(unsigned long long)(const void*)&Bs[0][0][0],
                  By, (unsigned)ldbt, (unsigned)K, 1u << 20);
  if (wid < 2) wait_tensor0();
#else
  copy_tile(As[0], Ay, lda, tid);
  copy_tile(Bs[0], By, ldbt, tid);
#endif
  __syncthreads();

  for (int kt = 0; kt < nk; ++kt) {
    const int cur = kt & 1, nxt = cur ^ 1;
    const bool more = (kt + 1) < nk;
    if (more) {
      const int k1 = (kt + 1) << 5;
#if USE_TDM
      if (wid == 0)
        tdm_load_tile((unsigned)(unsigned long long)(const void*)&As[nxt][0][0],
                      Ay + k1, (unsigned)lda, (unsigned)K, 1u << 20);
      else if (wid == 1)
        tdm_load_tile((unsigned)(unsigned long long)(const void*)&Bs[nxt][0][0],
                      By + k1, (unsigned)ldbt, (unsigned)K, 1u << 20);
#else
      copy_tile(As[nxt], Ay + k1, lda, tid);
      copy_tile(Bs[nxt], By + k1, ldbt, tid);
#endif
    }

    // ---- fragments (ISA 16-bit layouts; both tiles are [row][k]) ----
    union Frag { v16bf v; uint4 q[2]; } fa[2], fb[4];
#pragma unroll
    for (int tm = 0; tm < 2; ++tm) {
      int m = wm0 + tm * 16 + l15;
      fa[tm].q[0] = *(const uint4*)&As[cur][m][8 * kh];        // k 0..7 (+8h)
      fa[tm].q[1] = *(const uint4*)&As[cur][m][16 + 8 * kh];   // k 16..23 (+8h)
    }
#pragma unroll
    for (int tn = 0; tn < 4; ++tn) {
      int n = wn0 + tn * 16 + l15;
      fb[tn].q[0] = *(const uint4*)&Bs[cur][n][8 * kh];        // k 0..7 (+8h)
      fb[tn].q[1] = *(const uint4*)&Bs[cur][n][16 + 8 * kh];   // k 16..23 (+8h)
    }

#pragma unroll
    for (int tm = 0; tm < 2; ++tm)
#pragma unroll
      for (int tn = 0; tn < 4; ++tn)
        acc[tm][tn] = __builtin_amdgcn_wmma_f32_16x16x32_bf16(
            false, fa[tm].v, false, fb[tn].v, (short)0, acc[tm][tn],
            false, false);

#if USE_TDM
    if (more && wid < 2) wait_tensor0();
#endif
    __syncthreads();
  }

  // ------------------------------- epilogue --------------------------------
#pragma unroll
  for (int tm = 0; tm < 2; ++tm) {
#pragma unroll
    for (int tn = 0; tn < 4; ++tn) {
#pragma unroll
      for (int r = 0; r < 8; ++r) {
        int gm = bm0 + wm0 + tm * 16 + r + 8 * kh;   // global row
        int gn = bn0 + wn0 + tn * 16 + l15;          // global col
        float val = acc[tm][tn][r];

        if (mode == MODE_QKV) {
          // row = token b*S+s, col = h*128+d ; store [B,H,S,Dh] bf16 (+bias)
          float v2 = val + aux[gn];
          int b = gm >> 12, s = gm & 4095;
          int h = gn >> 7,  d = gn & 127;
          ((bf16_t*)Out)[(((long)(b * NH + h) * NS + s) << 7) + d] = f2bf(v2);
        } else if (mode == MODE_QKVT) {
          // V stored transposed: VT[bh][d][s]
          float v2 = val + aux[gn];
          int b = gm >> 12, s = gm & 4095;
          int h = gn >> 7,  d = gn & 127;
          ((bf16_t*)Out)[((long)(b * NH + h) * NDH + d) * NS + s] = f2bf(v2);
        } else if (mode == MODE_PHI_Q) {
          long base = ((long)blockIdx.y * NS + gm) * (long)NM;
          ((bf16_t*)Out)[base + gn]       = f2bf(__sinf(val));
          ((bf16_t*)Out)[base + 128 + gn] = f2bf(__cosf(val));
        } else if (mode == MODE_PHI_K) {
          // KfT[bh][m][s] (transposed so it is the A operand of the KV GEMM)
          long base = (long)blockIdx.y * NM * NS;
          ((bf16_t*)Out)[base + (long)gn * NS + gm]         = f2bf(__sinf(val));
          ((bf16_t*)Out)[base + (long)(gn + 128) * NS + gm] = f2bf(__cosf(val));
        } else if (mode == MODE_KVT) {
          // KV stored transposed: KVT[bh][d][m]
          ((bf16_t*)Out)[(long)blockIdx.y * NDH * NM + (long)gn * NM + gm] =
              f2bf(val);
        } else if (mode == MODE_OUT) {
          float dn = aux[(long)blockIdx.y * NS + gm];
          int b = blockIdx.y >> 3, h = blockIdx.y & 7;
          ((bf16_t*)Out)[((long)(b * NS + gm)) * ND + h * NDH + gn] =
              f2bf(val / dn);
        } else { // MODE_FINAL: fp32 out + bias
          ((float*)Out)[(long)gm * ND + gn] = val + aux[gn];
        }
      }
    }
  }
}

// --------------------- Ksum[bh,m] = sum_s KfT[bh,m,s] ----------------------
__global__ void ksum_kernel(const bf16_t* __restrict__ KfT,
                            float* __restrict__ Ksum) {
  int row = blockIdx.x;                       // bh*256 + m, 8192 rows
  const bf16_t* p = KfT + (long)row * NS;
  float s = 0.f;
  for (int i = threadIdx.x; i < NS; i += blockDim.x) s += (float)p[i];
  __shared__ float red[256];
  red[threadIdx.x] = s;
  __syncthreads();
  for (int off = 128; off; off >>= 1) {
    if ((int)threadIdx.x < off) red[threadIdx.x] += red[threadIdx.x + off];
    __syncthreads();
  }
  if (threadIdx.x == 0) Ksum[row] = red[0];
}

// ---------------- denom[bh,s] = Qf[bh,s,:] . Ksum[bh,:] + 1e-6 -------------
__global__ void denom_kernel(const bf16_t* __restrict__ Qf,
                             const float* __restrict__ Ksum,
                             float* __restrict__ denom) {
  long id = (long)blockIdx.x * blockDim.x + threadIdx.x;  // < B*H*S
  int bh = (int)(id >> 12);
  const bf16_t* q = Qf + id * NM;
  const float* ks = Ksum + (long)bh * NM;
  float s = 1e-6f;
#pragma unroll 8
  for (int i = 0; i < NM; ++i) s += (float)q[i] * ks[i];
  denom[id] = s;
}

// ---------------------------------------------------------------------------
extern "C" void kernel_launch(void* const* d_in, const int* in_sizes, int n_in,
                              void* d_out, int out_size, void* d_ws, size_t ws_size,
                              hipStream_t stream) {
  (void)in_sizes; (void)n_in; (void)out_size; (void)ws_size;

  const float* x     = (const float*)d_in[0];
  const float* Wq    = (const float*)d_in[1];
  const float* bq    = (const float*)d_in[2];
  const float* Wk    = (const float*)d_in[3];
  const float* bk    = (const float*)d_in[4];
  const float* Wv    = (const float*)d_in[5];
  const float* bv    = (const float*)d_in[6];
  const float* Wo    = (const float*)d_in[7];
  const float* bo    = (const float*)d_in[8];
  const float* omega = (const float*)d_in[9];

  // ---- workspace carve-up (256B aligned) ----
  char* w = (char*)d_ws;
  auto alloc = [&](size_t bytes) -> void* {
    void* p = (void*)w;
    w += (bytes + 255) & ~(size_t)255;
    return p;
  };
  const long nTok = (long)NB * NS;                 // 16384
  bf16_t* xb   = (bf16_t*)alloc(nTok * ND * 2);
  bf16_t* WqT  = (bf16_t*)alloc((long)ND * ND * 2);    // transposed weights
  bf16_t* WkT  = (bf16_t*)alloc((long)ND * ND * 2);
  bf16_t* WvT  = (bf16_t*)alloc((long)ND * ND * 2);
  bf16_t* WoT  = (bf16_t*)alloc((long)ND * ND * 2);
  bf16_t* omT  = (bf16_t*)alloc((long)NDH * NDH * 2);
  bf16_t* Qb   = (bf16_t*)alloc(nTok * ND * 2);        // [B,H,S,Dh]
  bf16_t* Kb   = (bf16_t*)alloc(nTok * ND * 2);        // [B,H,S,Dh]
  bf16_t* VTb  = (bf16_t*)alloc(nTok * ND * 2);        // [BH,Dh,S]
  bf16_t* Qf   = (bf16_t*)alloc((long)NB * NH * NS * NM * 2);  // [BH,S,M]
  bf16_t* KfT  = (bf16_t*)alloc((long)NB * NH * NM * NS * 2);  // [BH,M,S]
  float*  Ksum = (float*)alloc((long)NB * NH * NM * 4);
  bf16_t* KVT  = (bf16_t*)alloc((long)NB * NH * NDH * NM * 2); // [BH,Dh,M]
  float*  dnm  = (float*)alloc((long)NB * NH * NS * 4);
  bf16_t* Og   = (bf16_t*)alloc(nTok * ND * 2);        // [B,S,D]

  const int TPB = 256;

  // 1) fp32 -> bf16 (x straight; weights/omega transposed)
  cvt_kernel<<<4096, TPB, 0, stream>>>(x, xb, (int)(nTok * ND));
  dim3 gT(ND / 32, ND / 32);
  cvt_t_kernel<<<gT, TPB, 0, stream>>>(Wq, WqT, ND, ND);
  cvt_t_kernel<<<gT, TPB, 0, stream>>>(Wk, WkT, ND, ND);
  cvt_t_kernel<<<gT, TPB, 0, stream>>>(Wv, WvT, ND, ND);
  cvt_t_kernel<<<gT, TPB, 0, stream>>>(Wo, WoT, ND, ND);
  cvt_t_kernel<<<dim3(NDH / 32, NDH / 32), TPB, 0, stream>>>(omega, omT, NDH, NDH);

  // 2) QKV projections: [16384,1024] x [1024,1024]^T (+bias)
  dim3 gQKV((16384 / 128) * (1024 / 128), 1);
  wmma_gemm_kernel<<<gQKV, TPB, 0, stream>>>(xb, WqT, Qb, bq, ND, ND, 0, 0,
                                             ND, ND, MODE_QKV);
  wmma_gemm_kernel<<<gQKV, TPB, 0, stream>>>(xb, WkT, Kb, bk, ND, ND, 0, 0,
                                             ND, ND, MODE_QKV);
  wmma_gemm_kernel<<<gQKV, TPB, 0, stream>>>(xb, WvT, VTb, bv, ND, ND, 0, 0,
                                             ND, ND, MODE_QKVT);

  // 3) feature maps: per (b,h): [4096,128] x [128,128]^T -> sin||cos
  dim3 gPHI((4096 / 128) * (128 / 128), NB * NH);
  wmma_gemm_kernel<<<gPHI, TPB, 0, stream>>>(Qb, omT, Qf, nullptr, NDH, NDH,
                                             (long)NS * NDH, 0, NDH, NDH,
                                             MODE_PHI_Q);
  wmma_gemm_kernel<<<gPHI, TPB, 0, stream>>>(Kb, omT, KfT, nullptr, NDH, NDH,
                                             (long)NS * NDH, 0, NDH, NDH,
                                             MODE_PHI_K);

  // 4) Ksum
  ksum_kernel<<<NB * NH * NM, TPB, 0, stream>>>(KfT, Ksum);

  // 5) KV = KfT[256,4096] x VT[128,4096]^T per (b,h), stored transposed
  dim3 gKV((256 / 128) * (128 / 128), NB * NH);
  wmma_gemm_kernel<<<gKV, TPB, 0, stream>>>(KfT, VTb, KVT, nullptr, NS, NDH,
                                            (long)NM * NS, (long)NDH * NS,
                                            NS, NS, MODE_KVT);

  // 6) denom
  denom_kernel<<<(NB * NH * NS) / TPB, TPB, 0, stream>>>(Qf, Ksum, dnm);

  // 7) out = Qf[4096,256] x KVT[128,256]^T / denom, relayout to [B,S,D]
  dim3 gOUT((4096 / 128) * (128 / 128), NB * NH);
  wmma_gemm_kernel<<<gOUT, TPB, 0, stream>>>(Qf, KVT, Og, dnm, NM, NDH,
                                             (long)NS * NM, (long)NDH * NM,
                                             NM, NM, MODE_OUT);

  // 8) final: [16384,1024] x [1024,1024]^T + bo -> fp32 d_out
  dim3 gFIN((16384 / 128) * (1024 / 128), 1);
  wmma_gemm_kernel<<<gFIN, TPB, 0, stream>>>(Og, WoT, d_out, bo, ND, ND, 0, 0,
                                             ND, ND, MODE_FINAL);
}
